// AlibiAttentionBlock_30829275250973
// MI455X (gfx1250) — matrix operand order repaired and checked by reading, not compiled, not run on hardware
//
#include <hip/hip_runtime.h>
#include <hip/hip_bf16.h>
#include <math.h>

// ---------------------------------------------------------------------------
// CDNA5 (gfx1250) bf16 WMMA helpers.  Wave32 only.
// ---------------------------------------------------------------------------
typedef __attribute__((ext_vector_type(16))) __bf16 v16bf;
typedef __attribute__((ext_vector_type(8)))  float  v8f;
typedef __attribute__((ext_vector_type(4)))  unsigned int u32x4;
typedef __attribute__((ext_vector_type(8)))  int    i32x8;
typedef __attribute__((ext_vector_type(4)))  int    i32x4;

#define LOG2E 1.44269504088896340736f

__device__ __forceinline__ v8f wmma_bf16(v16bf a, v16bf b, v8f c) {
    // D = A(16x32) * B(32x16) + C(16x16), f32 accumulate
    return __builtin_amdgcn_wmma_f32_16x16x32_bf16(
        /*neg_a=*/false, a, /*neg_b=*/false, b,
        /*c_mod=*/(short)0, c, /*reuse_a=*/false, /*reuse_b=*/false);
}

// A fragment: 16x32 bf16, row-major source with element stride `stride`.
// ISA layout: lanes 0-15 rows M=0..15; VGPR j (j<4) holds K=half*8+2j,2j+1,
// VGPR 4+j holds K=16+half*8+2j,..., where half = lane>>4.
__device__ __forceinline__ v16bf load_a_frag(const __bf16* base, unsigned stride, int lane) {
    const int m = lane & 15, half = lane >> 4;
    const __bf16* row = base + (unsigned)m * stride;
    v16bf a;
#pragma unroll
    for (int j = 0; j < 4; ++j) {
        const int k0 = half * 8 + 2 * j;
        a[2*j]       = row[k0];
        a[2*j+1]     = row[k0 + 1];
        a[8 + 2*j]   = row[16 + k0];
        a[8 + 2*j+1] = row[16 + k0 + 1];
    }
    return a;
}

// B fragment: 32x16 bf16 read from a TRANSPOSED staging buffer Bt[n][k].
// ISA layout: VGPR j (j<4) holds K=half*8+2j,2j+1, VGPR 4+j holds K=16+half*8+2j,...
__device__ __forceinline__ v16bf load_b_frag_t(const __bf16* baseT, unsigned stride, int lane) {
    const int n = lane & 15, half = lane >> 4;
    const __bf16* col = baseT + (unsigned)n * stride;
    v16bf b;
#pragma unroll
    for (int j = 0; j < 8; ++j) {
        const int k0 = half * 8 + 2 * (j & 3) + (j >> 2) * 16;
        b[2*j]   = col[k0];
        b[2*j+1] = col[k0 + 1];
    }
    return b;
}

// ---------------------------------------------------------------------------
// Tensor Data Mover: async 2-D bf16 tile load (global -> LDS).
// Descriptor packing per CDNA5 ISA ch.8 (D# group0/group1 bit layout).
// This toolchain exposes the 6-arg builtin (amdgpu-toolchain/clang-23 form):
//   (uint32x4 g0, int32x8 g1, int32x4 g2, int32x4 g3, int32x8 pad, i32 cpol)
// ---------------------------------------------------------------------------
__device__ __forceinline__ void tdm_load_2d_bf16(
    unsigned lds_addr, const void* gaddr,
    unsigned tile_d0, unsigned tile_d1,
    unsigned tensor_d0, unsigned tensor_d1,
    unsigned stride0_elems)
{
    const unsigned long long ga = (unsigned long long)(uintptr_t)gaddr;
    u32x4 g0;
    g0[0] = 1u;                                   // count=1, user descriptor
    g0[1] = lds_addr;                             // LDS byte address
    g0[2] = (unsigned)(ga & 0xffffffffull);       // global_addr[31:0]
    g0[3] = (unsigned)(ga >> 32) | (2u << 30);    // global_addr[56:32] | type=2
    i32x8 g1;
    g1[0] = (int)(1u << 16);                      // data_size=1 (2-byte elems)
    g1[1] = (int)((tensor_d0 & 0xffffu) << 16);   // tensor_dim0[15:0] @ [63:48]
    g1[2] = (int)((tensor_d0 >> 16) | ((tensor_d1 & 0xffffu) << 16));
    g1[3] = (int)((tensor_d1 >> 16) | (tile_d0 << 16));   // tile_dim0 @ [127:112]
    g1[4] = (int)(tile_d1 & 0xffffu);             // tile_dim1; tile_dim2=0 (2D)
    g1[5] = (int)stride0_elems;                   // tensor_dim0_stride[31:0]
    g1[6] = 0;                                    // stride0[47:32] | stride1[15:0]
    g1[7] = 0;
    const i32x4 z4 = {0, 0, 0, 0};
    const i32x8 z8 = {0, 0, 0, 0, 0, 0, 0, 0};
    __builtin_amdgcn_tensor_load_to_lds(g0, g1, z4, z4, z8, 0);
}

// ---------------------------------------------------------------------------
// f32 -> bf16 conversion (one-time cost; makes all GEMMs WMMA-eligible)
// ---------------------------------------------------------------------------
__global__ void cvt_f32_bf16(const float* __restrict__ in, __bf16* __restrict__ out, int n) {
    int i = (blockIdx.x * blockDim.x + threadIdx.x) * 4;
    if (i + 3 < n) {
        float4 v = *(const float4*)(in + i);
        out[i+0] = (__bf16)v.x;
        out[i+1] = (__bf16)v.y;
        out[i+2] = (__bf16)v.z;
        out[i+3] = (__bf16)v.w;
    } else {
        for (; i < n; ++i) out[i] = (__bf16)in[i];
    }
}

// ---------------------------------------------------------------------------
// Generic bf16 WMMA GEMM:  out[M,N] = A[M,K] * B[K,N] + bias[N]
// Block tile 64x128, 256 threads = 8 waves in 2x4 grid, wave tile 32x32.
// Requires M%64==0, N%128==0, K%32==0.  All offsets 32-bit (buffers < 2^31).
// ---------------------------------------------------------------------------
template <bool OUT_BF16>
__global__ __launch_bounds__(256) void gemm_bf16_kernel(
    const __bf16* __restrict__ A, const __bf16* __restrict__ B,
    const float* __restrict__ bias, void* __restrict__ outp,
    int M, int N, int K)
{
    __shared__ __bf16 sA[64 * 32];    // A tile, row-major
    __shared__ __bf16 sBt[128 * 32];  // B tile, transposed: sBt[n][k]

    const int tid  = threadIdx.x;
    const int lane = tid & 31;
    const int wave = tid >> 5;
    const int wm   = wave >> 2;   // 0..1
    const int wn   = wave & 3;    // 0..3
    const unsigned bm = blockIdx.y * 64;
    const unsigned bn = blockIdx.x * 128;

    v8f acc[2][2];
#pragma unroll
    for (int i = 0; i < 2; ++i)
#pragma unroll
        for (int j = 0; j < 2; ++j)
#pragma unroll
            for (int r = 0; r < 8; ++r) acc[i][j][r] = 0.0f;

    const unsigned ar = tid >> 2, ac = (tid & 3) * 8;   // A loader: 16B/thread
    const unsigned bk = tid >> 3, bc = (tid & 7) * 16;  // B loader: 32B/thread

    for (unsigned k0 = 0; k0 < (unsigned)K; k0 += 32) {
        *(uint4*)(&sA[ar * 32 + ac]) =
            *(const uint4*)(&A[(bm + ar) * (unsigned)K + k0 + ac]);

        union { uint4 u[2]; __bf16 h[16]; } tb;
        const unsigned boff = (k0 + bk) * (unsigned)N + bn + bc;
        tb.u[0] = *(const uint4*)(&B[boff]);
        tb.u[1] = *(const uint4*)(&B[boff + 8]);
#pragma unroll
        for (int i = 0; i < 16; ++i) sBt[(bc + i) * 32 + bk] = tb.h[i];
        __syncthreads();

        v16bf af[2], bf[2];
#pragma unroll
        for (int i = 0; i < 2; ++i)
            af[i] = load_a_frag(&sA[(wm * 32 + i * 16) * 32], 32, lane);
#pragma unroll
        for (int j = 0; j < 2; ++j)
            bf[j] = load_b_frag_t(&sBt[(wn * 32 + j * 16) * 32], 32, lane);
#pragma unroll
        for (int i = 0; i < 2; ++i)
#pragma unroll
            for (int j = 0; j < 2; ++j)
                acc[i][j] = wmma_bf16(af[i], bf[j], acc[i][j]);
        __syncthreads();
    }

    // Epilogue: branch-free, 32-bit addressing, bias hoisted per column.
    const int nl = lane & 15, half = lane >> 4;
    const unsigned mbase = bm + wm * 32 + half * 8;
    const unsigned nbase = bn + wn * 32 + nl;
#pragma unroll
    for (int j = 0; j < 2; ++j) {
        const unsigned n = nbase + j * 16;
        const float bj = bias[n];
#pragma unroll
        for (int i = 0; i < 2; ++i) {
#pragma unroll
            for (int r = 0; r < 8; ++r) {
                const unsigned off = (mbase + i * 16 + r) * (unsigned)N + n;
                const float v = acc[i][j][r] + bj;
                if (OUT_BF16) ((__bf16*)outp)[off] = (__bf16)v;
                else          ((float*)outp)[off]  = v;
            }
        }
    }
}

// ---------------------------------------------------------------------------
// Fused ALiBi flash attention.
// qkv: bf16 [B*L=4096, 3072] = [q | k | v], each [16 heads x 64 dims].
// O:   bf16 [4096, 1024].
// grid = (L/64, B*H), block = 128 threads = 4 waves; wave owns 16 q-rows.
// K tile arrives via the Tensor Data Mover (async DMA into LDS).
// ---------------------------------------------------------------------------
__global__ __launch_bounds__(128) void alibi_attn_kernel(
    const __bf16* __restrict__ qkv, __bf16* __restrict__ O)
{
    __shared__ __bf16 sK [64 * 64];      // keys x dims (row-major == Bt for QK^T)
    __shared__ __bf16 sVt[64 * 64];      // dims x keys (transposed V)
    __shared__ __bf16 sP [4 * 16 * 64];  // per-wave P tile (C-layout -> A-layout)

    const int tid  = threadIdx.x;
    const int lane = tid & 31;
    const int wave = tid >> 5;             // 0..3
    const int nl   = lane & 15, half = lane >> 4;

    const int bh = blockIdx.y;
    const int b  = bh >> 4, h = bh & 15;
    const int q0 = blockIdx.x * 64;
    // Logits are pre-scaled by log2(e) so softmax uses a single v_exp_f32
    // (base-2) per element: exp(x-m) == exp2(x*log2e - m*log2e).
    const float qscale = 0.125f * LOG2E;                       // 1/sqrt(64)*log2e
    const float slope2 = exp2f(-0.5f * (float)h) * LOG2E;      // ALiBi slope*log2e

    // Q fragments stay in registers for the whole kernel (hd=64 -> 2 K-chunks)
    const __bf16* Qb = qkv + (unsigned)(b * 2048 + q0 + wave * 16) * 3072u + h * 64u;
    const v16bf aq0 = load_a_frag(Qb,      3072, lane);
    const v16bf aq1 = load_a_frag(Qb + 32, 3072, lane);

    float rmax[8], rsum[8];
    v8f oacc[4];
#pragma unroll
    for (int r = 0; r < 8; ++r) { rmax[r] = -1e30f; rsum[r] = 0.0f; }
#pragma unroll
    for (int t = 0; t < 4; ++t)
#pragma unroll
        for (int r = 0; r < 8; ++r) oacc[t][r] = 0.0f;

    const int ldrow = tid >> 1, ldc = (tid & 1) * 32;   // V-tile loader
    const int qrow_base = q0 + wave * 16 + half * 8;
    const unsigned kv_base = (unsigned)(b * 2048) * 3072u + h * 64u;

    for (int kb = 0; kb < 32; ++kb) {
        // K tile (64 keys x 64 dims): single async TDM transfer, wave 0 only
        // (tensor ops ignore EXEC; the uniform branch keeps it to one issue).
        if (wave == 0) {
            tdm_load_2d_bf16((unsigned)(uintptr_t)&sK[0],
                             qkv + kv_base + (unsigned)(kb * 64) * 3072u + 1024u,
                             /*tile=*/64, 64,
                             /*tensor=*/64, (unsigned)(2048 - kb * 64),
                             /*stride0=*/3072);
        }
        {   // V tile, stored transposed: sVt[d][key]
            const __bf16* src = qkv + kv_base + (unsigned)(kb * 64 + ldrow) * 3072u
                                    + 2048u + ldc;
            const uint4* s4 = (const uint4*)src;
            union { uint4 u[4]; __bf16 hh[32]; } tv;
            tv.u[0] = s4[0]; tv.u[1] = s4[1]; tv.u[2] = s4[2]; tv.u[3] = s4[3];
#pragma unroll
            for (int i = 0; i < 32; ++i) sVt[(ldc + i) * 64 + ldrow] = tv.hh[i];
        }
        if (wave == 0) __builtin_amdgcn_s_wait_tensorcnt(0);
        __syncthreads();

        // S = (Q K^T): 4 key-tiles of 16, each = 2 chained WMMAs over hd
        v8f s[4];
#pragma unroll
        for (int j = 0; j < 4; ++j) {
            v8f z;
#pragma unroll
            for (int r = 0; r < 8; ++r) z[r] = 0.0f;
            const v16bf bk0 = load_b_frag_t(&sK[j*16*64],      64, lane);
            const v16bf bk1 = load_b_frag_t(&sK[j*16*64 + 32], 64, lane);
            z    = wmma_bf16(aq0, bk0, z);
            s[j] = wmma_bf16(aq1, bk1, z);
        }

        // Online softmax with ALiBi bias (all in base-2 logit domain)
#pragma unroll
        for (int r = 0; r < 8; ++r) {
            float mr = -1e30f;
#pragma unroll
            for (int j = 0; j < 4; ++j) {
                const float kpos = (float)(kb*64 + j*16 + nl);
                const float val = s[j][r] * qscale
                                - slope2 * fabsf((float)(qrow_base + r) - kpos);
                s[j][r] = val;
                mr = fmaxf(mr, val);
            }
#pragma unroll
            for (int off = 1; off < 16; off <<= 1)       // reduce across the 16
                mr = fmaxf(mr, __shfl_xor(mr, off, 32)); // lanes of this half
            const float nm    = fmaxf(rmax[r], mr);
            const float alpha = exp2f(rmax[r] - nm);
            rmax[r] = nm;
            float ps = 0.0f;
#pragma unroll
            for (int j = 0; j < 4; ++j) {
                const float p = exp2f(s[j][r] - nm);
                s[j][r] = p;
                ps += p;
            }
#pragma unroll
            for (int off = 1; off < 16; off <<= 1)
                ps += __shfl_xor(ps, off, 32);
            rsum[r] = rsum[r] * alpha + ps;
#pragma unroll
            for (int t = 0; t < 4; ++t) oacc[t][r] *= alpha;
            const int m = half * 8 + r;                  // C-layout -> LDS row-major
#pragma unroll
            for (int j = 0; j < 4; ++j)
                sP[wave*1024 + m*64 + j*16 + nl] = (__bf16)s[j][r];
        }

        // O += P * V   (P via per-wave LDS round-trip; in-wave DS ops in order)
        const v16bf pa0 = load_a_frag(&sP[wave*1024],      64, lane);
        const v16bf pa1 = load_a_frag(&sP[wave*1024 + 32], 64, lane);
#pragma unroll
        for (int t = 0; t < 4; ++t) {
            const v16bf bv0 = load_b_frag_t(&sVt[t*16*64],      64, lane);
            const v16bf bv1 = load_b_frag_t(&sVt[t*16*64 + 32], 64, lane);
            oacc[t] = wmma_bf16(pa0, bv0, oacc[t]);
            oacc[t] = wmma_bf16(pa1, bv1, oacc[t]);
        }
        __syncthreads();
    }

    // Normalize and write O (bf16, feeds the final WMMA projection)
#pragma unroll
    for (int r = 0; r < 8; ++r) {
        const unsigned tok = (unsigned)(b * 2048 + qrow_base + r);
        const float inv = 1.0f / rsum[r];
#pragma unroll
        for (int t = 0; t < 4; ++t)
            O[tok * 1024u + h*64u + t*16u + nl] = (__bf16)(oacc[t][r] * inv);
    }
}

// ---------------------------------------------------------------------------
// Launch: cvt x3  ->  QKV GEMM  ->  fused attention  ->  output GEMM
// Workspace layout (48 MB total):
//   [0,8M)  x_bf16   [8M,14M) Wqkv_bf16   [14M,38M) qkv_bf16
//   [38M,46M) O_bf16   [46M,48M) Wout_bf16
// ---------------------------------------------------------------------------
extern "C" void kernel_launch(void* const* d_in, const int* in_sizes, int n_in,
                              void* d_out, int out_size, void* d_ws, size_t ws_size,
                              hipStream_t stream) {
    const float* x    = (const float*)d_in[0];
    const float* Wqkv = (const float*)d_in[1];
    const float* bqkv = (const float*)d_in[2];
    const float* Wout = (const float*)d_in[3];
    const float* bout = (const float*)d_in[4];

    const int BL = 4096, D = 1024, N3 = 3072;

    char* ws = (char*)d_ws;
    __bf16* xb    = (__bf16*)(ws);
    __bf16* wqkvb = (__bf16*)(ws + (size_t)( 8u << 20));
    __bf16* qkvb  = (__bf16*)(ws + (size_t)(14u << 20));
    __bf16* ob    = (__bf16*)(ws + (size_t)(38u << 20));
    __bf16* woutb = (__bf16*)(ws + (size_t)(46u << 20));

    const int n1 = BL * D, n2 = D * N3, n3 = D * D;
    cvt_f32_bf16<<<(n1/4 + 255)/256, 256, 0, stream>>>(x,    xb,    n1);
    cvt_f32_bf16<<<(n2/4 + 255)/256, 256, 0, stream>>>(Wqkv, wqkvb, n2);
    cvt_f32_bf16<<<(n3/4 + 255)/256, 256, 0, stream>>>(Wout, woutb, n3);

    // qkv = x @ Wqkv + bqkv   (bf16 out, stays in L2)
    gemm_bf16_kernel<true><<<dim3(N3/128, BL/64), 256, 0, stream>>>(
        xb, wqkvb, bqkv, qkvb, BL, N3, D);

    // fused ALiBi flash attention
    alibi_attn_kernel<<<dim3(32, 32), 128, 0, stream>>>(qkvb, ob);

    // out = O @ Wout + bout   (f32 out)
    gemm_bf16_kernel<false><<<dim3(D/128, BL/64), 256, 0, stream>>>(
        ob, woutb, bout, d_out, BL, D, D);
}
